// ProductLayer_53549652246913
// MI455X (gfx1250) — matrix operand order repaired and checked by reading, not compiled, not run on hardware
//
#include <hip/hip_runtime.h>
#include <cstdint>
#include <cstddef>

// ---------------------------------------------------------------------------
// Problem constants (from reference)
// ---------------------------------------------------------------------------
#define NN     10000
#define EE     40000
#define SS     64
#define BB     32
#define CC     128
#define GRp    12
#define GFp    24
#define FF     160           // 2S+B
#define D1     3072          // C*GF
#define D3     768           // S*GR
#define EPS    1e-5f

#define MTILE  128           // GEMM M tile
#define MPAD   41600         // EE bucket-padded to 128, upper bound (= 5*8320)
#define MTILES (MPAD/128)    // 325
#define CHUNK  8320          // rows per M-chunk (65 * 128)
#define NCHUNK (MPAD/CHUNK)  // 5
#define LDS_STR 40           // LDS row stride (halves), 80B = bank-conflict pad

typedef __attribute__((ext_vector_type(16))) _Float16 v16h;
typedef __attribute__((ext_vector_type(8)))  float    v8f;

union Frag16 { v16h h; uint4 q[2]; };

// ---------------------------------------------------------------------------
// CDNA5 async global->LDS copy (ASYNCcnt-tracked, no VGPR round trip).
// vdst = per-lane LDS byte address, vaddr = per-lane 64-bit global address.
// LDS aperture keeps the byte offset in the low 32 bits of a generic pointer
// (ISA 10.2 per-aperture address calc), so a truncating cast is sufficient.
// ---------------------------------------------------------------------------
__device__ __forceinline__ void async_ld_b128(const _Float16* lds_ptr, const _Float16* g) {
    uint32_t lds_off = (uint32_t)(uintptr_t)lds_ptr;
    asm volatile("global_load_async_to_lds_b128 %0, %1, off"
                 :: "v"(lds_off), "v"(g)
                 : "memory");
}
__device__ __forceinline__ void wait_async_le4() {
    asm volatile("s_wait_asynccnt 0x4" ::: "memory");
}
__device__ __forceinline__ void wait_async_0() {
    asm volatile("s_wait_asynccnt 0x0" ::: "memory");
}

// ---------------------------------------------------------------------------
// Utility kernels
// ---------------------------------------------------------------------------
__global__ void fill_u32(uint32_t* p, uint32_t v, long long n) {
    long long i = (long long)blockIdx.x * blockDim.x + threadIdx.x;
    if (i < n) p[i] = v;
}

__global__ void cast_f32_f16(const float* __restrict__ src, _Float16* __restrict__ dst, long long n) {
    long long i = (long long)blockIdx.x * blockDim.x + threadIdx.x;
    if (i < n) dst[i] = (_Float16)src[i];
}

// W1 [Cout=128, Gout=24, Cin=160, Gin=12] -> W1t[g][n = o*24+h][c] (f16)
__global__ void convert_w1(const float* __restrict__ w, _Float16* __restrict__ wt) {
    long long i = (long long)blockIdx.x * blockDim.x + threadIdx.x;
    const long long total = (long long)GRp * D1 * FF;
    if (i >= total) return;
    int c   = (int)(i % FF);
    int n   = (int)((i / FF) % D1);
    int g   = (int)(i / ((long long)FF * D1));
    wt[i] = (_Float16)w[((long long)n * FF + c) * GRp + g];
}

// g index per edge (argmax of one-hot) + histogram
__global__ void compute_g_hist(const float* __restrict__ oh, int* __restrict__ gOf,
                               int* __restrict__ counts) {
    int e = blockIdx.x * blockDim.x + threadIdx.x;
    if (e >= EE) return;
    const float* o = oh + (long long)e * GRp;
    int g = 0; float best = o[0];
    for (int j = 1; j < GRp; j++) { float v = o[j]; if (v > best) { best = v; g = j; } }
    gOf[e] = g;
    atomicAdd(&counts[g], 1);
}

// Single-thread: 128-aligned bucket starts, per-M-tile group table, cursors
__global__ void scan_tiles(const int* __restrict__ counts, int* __restrict__ starts,
                           int* __restrict__ tileG, int* __restrict__ cursor) {
    if (threadIdx.x != 0 || blockIdx.x != 0) return;
    int s = 0;
    for (int g = 0; g < GRp; g++) {
        starts[g] = s;
        cursor[g] = s;
        s = (s + counts[g] + 127) & ~127;
    }
    starts[GRp] = s;
    for (int t = 0; t < MTILES; t++) {
        int row = t * 128; int g = 0;
        for (int q = 0; q < GRp; q++)
            if (row >= starts[q] && row < starts[q + 1]) g = q;
        tileG[t] = g;
    }
}

__global__ void scatter_rows(const int* __restrict__ gOf, int* __restrict__ cursor,
                             int* __restrict__ rowOfEdge, int* __restrict__ edgeOfRow) {
    int e = blockIdx.x * blockDim.x + threadIdx.x;
    if (e >= EE) return;
    int pos = atomicAdd(&cursor[gOf[e]], 1);
    rowOfEdge[e] = pos;
    edgeOfRow[pos] = e;
}

// Build A1[pos, 0:160] = [sites[i1], sites[i2], bonds]  (f16, bucket-permuted)
__global__ void build_A1(const float* __restrict__ sites, const int* __restrict__ i1,
                         const int* __restrict__ i2, const float* __restrict__ bonds,
                         const int* __restrict__ rowOfEdge, _Float16* __restrict__ A1) {
    int e = blockIdx.x, t = threadIdx.x;          // grid EE x 160
    int pos = rowOfEdge[e];
    float v;
    if (t < SS)           v = sites[(long long)i1[e] * SS + t];
    else if (t < 2 * SS)  v = sites[(long long)i2[e] * SS + (t - SS)];
    else                  v = bonds[(long long)e * BB + (t - 2 * SS)];
    A1[(long long)pos * FF + t] = (_Float16)v;
}

// ---------------------------------------------------------------------------
// WMMA GEMM: D[m,n] = sum_k A[m,k] * Bt[n,k]  (f16 in, f32 acc, f16 out)
// Block: 256 thr (8 waves), tile 128x128, K-step 32.
// Per wave: 32x64 region = 2x4 subtiles of v_wmma_f32_16x16x32_f16.
// Staging: double-buffered LDS filled by global_load_async_to_lds_b128
// (4 async b128 copies per thread per K-step -> constant s_wait_asynccnt 4
// pipeline wait; async loads complete in order so the oldest 4 belong to the
// buffer about to be consumed).
// tileG != nullptr selects a per-M-tile weight block (layer-1 group weights).
// Fragment layouts per CDNA5 ISA 7.12.2 (16-bit A 16x32 / f32 C-D 16x16).
// ---------------------------------------------------------------------------
__global__ __launch_bounds__(256)
void gemm_wmma_f16(const _Float16* __restrict__ A, int lda,
                   const _Float16* __restrict__ Bt,
                   _Float16* __restrict__ D, int ldd, int K,
                   const int* __restrict__ tileG, int mTile0, long long bStrideG) {
    __shared__ _Float16 sA[2][128 * LDS_STR];
    __shared__ _Float16 sB[2][128 * LDS_STR];

    const int tid  = threadIdx.x;
    const int lane = tid & 31;
    const int wave = tid >> 5;
    const int wm   = (wave & 3) * 32;
    const int wn   = (wave >> 2) * 64;

    const long long mBase = (long long)blockIdx.y * 128;
    const long long nBase = (long long)blockIdx.x * 128;

    const _Float16* Bbase = Bt;
    if (tileG) Bbase += (long long)tileG[mTile0 + blockIdx.y] * bStrideG;

    // staging coordinates for this thread (2 b128 chunks for A, 2 for B)
    const int idx0 = tid * 2;
    const int row0 = idx0 >> 2,      kc0 = (idx0 & 3) * 8;
    const int row1 = (idx0 + 1) >> 2, kc1 = ((idx0 + 1) & 3) * 8;

    v8f acc[2][4];
    const v8f vzero = {0.f, 0.f, 0.f, 0.f, 0.f, 0.f, 0.f, 0.f};
    for (int i = 0; i < 2; i++)
        for (int j = 0; j < 4; j++) acc[i][j] = vzero;

    const int nk = K / 32;

    // prologue: stage tile 0 into buffer 0
    {
        const int k0 = 0;
        async_ld_b128(&sA[0][row0 * LDS_STR + kc0], A + (mBase + row0) * (long long)lda + k0 + kc0);
        async_ld_b128(&sB[0][row0 * LDS_STR + kc0], Bbase + (nBase + row0) * (long long)K + k0 + kc0);
        async_ld_b128(&sA[0][row1 * LDS_STR + kc1], A + (mBase + row1) * (long long)lda + k0 + kc1);
        async_ld_b128(&sB[0][row1 * LDS_STR + kc1], Bbase + (nBase + row1) * (long long)K + k0 + kc1);
    }

    for (int kt = 0; kt < nk; kt++) {
        const int cur = kt & 1;
        if (kt + 1 < nk) {
            // prefetch next K-tile into the other buffer while we compute
            const int nxt = cur ^ 1;
            const int k0  = (kt + 1) * 32;
            async_ld_b128(&sA[nxt][row0 * LDS_STR + kc0], A + (mBase + row0) * (long long)lda + k0 + kc0);
            async_ld_b128(&sB[nxt][row0 * LDS_STR + kc0], Bbase + (nBase + row0) * (long long)K + k0 + kc0);
            async_ld_b128(&sA[nxt][row1 * LDS_STR + kc1], A + (mBase + row1) * (long long)lda + k0 + kc1);
            async_ld_b128(&sB[nxt][row1 * LDS_STR + kc1], Bbase + (nBase + row1) * (long long)K + k0 + kc1);
            wait_async_le4();   // oldest 4 (current buffer) have landed in LDS
        } else {
            wait_async_0();     // drain
        }
        __syncthreads();        // current buffer visible to whole block

        v16h af[2], bf[4];
        {
            const int r  = lane & 15;
            const int ka = (lane < 16) ? 0 : 8;    // A: lanes16-31 hold K 8..15,24..31
            for (int i = 0; i < 2; i++) {
                const _Float16* s = &sA[cur][(wm + i * 16 + r) * LDS_STR];
                Frag16 f;
                f.q[0] = *(const uint4*)(s + ka);        // K = ka .. ka+7
                f.q[1] = *(const uint4*)(s + ka + 16);   // K = ka+16 .. ka+23
                af[i] = f.h;
            }
            const int kb = (lane < 16) ? 0 : 8;    // B: lanes16-31 hold K 8..15,24..31
            for (int j = 0; j < 4; j++) {
                const _Float16* s = &sB[cur][(wn + j * 16 + r) * LDS_STR];
                Frag16 f;
                f.q[0] = *(const uint4*)(s + kb);
                f.q[1] = *(const uint4*)(s + kb + 16);
                bf[j] = f.h;
            }
        }
        for (int i = 0; i < 2; i++)
            for (int j = 0; j < 4; j++)
                acc[i][j] = __builtin_amdgcn_wmma_f32_16x16x32_f16(
                    false, af[i], false, bf[j], (short)0, acc[i][j], false, false);

        __syncthreads();        // all reads of 'cur' done before it is re-staged
    }

    // epilogue: C/D layout — VGPR r: lanes0-15 M=r, lanes16-31 M=r+8; N=lane&15
    const int mOff = (lane >= 16) ? 8 : 0;
    const int nOff = lane & 15;
    for (int i = 0; i < 2; i++)
        for (int j = 0; j < 4; j++)
            for (int r = 0; r < 8; r++) {
                long long m = mBase + wm + i * 16 + r + mOff;
                long long n = nBase + wn + j * 16 + nOff;
                D[m * (long long)ldd + n] = (_Float16)acc[i][j][r];
            }
}

// ---------------------------------------------------------------------------
// Row LayerNorm (+bias before stats, +ReLU), in place on f16 rows. grid = rows.
// ---------------------------------------------------------------------------
__global__ __launch_bounds__(256)
void ln_rows(_Float16* __restrict__ buf, int L, const float* __restrict__ bias,
             const float* __restrict__ gamma, const float* __restrict__ beta) {
    __shared__ float rs[256], rq[256];
    const int t = threadIdx.x;
    const long long row = blockIdx.x;
    const int per = L / 256;                 // 12 (D1) or 3 (D3)
    float loc[12];
    float s = 0.f, q = 0.f;
    for (int i = 0; i < per; i++) {
        int idx = i * 256 + t;
        float x = (float)buf[row * L + idx] + bias[idx];
        loc[i] = x; s += x; q += x * x;
    }
    rs[t] = s; rq[t] = q; __syncthreads();
    for (int off = 128; off > 0; off >>= 1) {
        if (t < off) { rs[t] += rs[t + off]; rq[t] += rq[t + off]; }
        __syncthreads();
    }
    float mean = rs[0] / (float)L;
    float var  = rq[0] / (float)L - mean * mean;
    float inv  = rsqrtf(var + EPS);
    for (int i = 0; i < per; i++) {
        int idx = i * 256 + t;
        float x = (loc[i] - mean) * inv * gamma[idx] + beta[idx];
        buf[row * L + idx] = (_Float16)fmaxf(x, 0.f);
    }
}

// ---------------------------------------------------------------------------
// Gather column i1_cells, apply constant gate (LN over 1 elem == beta, so
// gate = sigmoid(attn_beta)), atomic segment-sum into upd[N,S].
// ---------------------------------------------------------------------------
__global__ void gate_scatter(const _Float16* __restrict__ Y3, const int* __restrict__ edgeOfRow,
                             int m0, const int* __restrict__ i1c, const int* __restrict__ i1,
                             const float* __restrict__ attnBeta, float* __restrict__ upd) {
    int rl = blockIdx.x * blockDim.y + threadIdx.y;   // local row in chunk
    if (rl >= CHUNK) return;
    int e = edgeOfRow[m0 + rl];
    if (e < 0) return;
    int s = threadIdx.x;                               // 0..63
    int k = i1c[e];
    float ls   = (float)Y3[(long long)rl * D3 + s * GRp + k];
    float gate = 1.f / (1.f + expf(-attnBeta[0]));
    atomicAdd(&upd[(long long)i1[e] * SS + s], gate * ls);
}

// ---------------------------------------------------------------------------
// Node MLP: v=[upd,sites] (128) -> fc1(64)+LN+ReLU -> fc2(64)+LN -> +sites
// one block (64 threads) per node
// ---------------------------------------------------------------------------
__global__ __launch_bounds__(64)
void node_update(const float* __restrict__ sites, const float* __restrict__ upd,
                 const float* __restrict__ w1, const float* __restrict__ b1,
                 const float* __restrict__ g1, const float* __restrict__ be1,
                 const float* __restrict__ w2, const float* __restrict__ b2,
                 const float* __restrict__ g2, const float* __restrict__ be2,
                 float* __restrict__ out) {
    __shared__ float v[128], h[64], rs[64], rq[64];
    const int n = blockIdx.x, t = threadIdx.x;
    v[t]      = upd[(long long)n * SS + t];
    v[SS + t] = sites[(long long)n * SS + t];
    __syncthreads();

    float y = b1[t];
    for (int j = 0; j < 128; j++) y += v[j] * w1[t * 128 + j];
    rs[t] = y; rq[t] = y * y; __syncthreads();
    for (int off = 32; off > 0; off >>= 1) {
        if (t < off) { rs[t] += rs[t + off]; rq[t] += rq[t + off]; }
        __syncthreads();
    }
    float mean = rs[0] / 64.f, var = rq[0] / 64.f - (rs[0] / 64.f) * (rs[0] / 64.f);
    float x = (y - mean) * rsqrtf(var + EPS) * g1[t] + be1[t];
    x = fmaxf(x, 0.f);
    h[t] = x; __syncthreads();

    float y2 = b2[t];
    for (int j = 0; j < 64; j++) y2 += h[j] * w2[t * 64 + j];
    rs[t] = y2; rq[t] = y2 * y2; __syncthreads();
    for (int off = 32; off > 0; off >>= 1) {
        if (t < off) { rs[t] += rs[t + off]; rq[t] += rq[t + off]; }
        __syncthreads();
    }
    mean = rs[0] / 64.f; var = rq[0] / 64.f - mean * mean;
    float o = (y2 - mean) * rsqrtf(var + EPS) * g2[t] + be2[t];
    out[(long long)n * SS + t] = sites[(long long)n * SS + t] + o;
}

// ---------------------------------------------------------------------------
// Host launch
// ---------------------------------------------------------------------------
struct StreamP {
    const float *bonds, *oh;
    const float *w1, *b1, *be1, *g1;
    const float *w2, *b2, *be2, *g2;
    const float *w3, *b3, *be3, *g3;
    const float *attnBeta;
    const int *i1, *i2, *i1c;
};

extern "C" void kernel_launch(void* const* d_in, const int* in_sizes, int n_in,
                              void* d_out, int out_size, void* d_ws, size_t ws_size,
                              hipStream_t stream) {
    // Input order: setup_inputs() insertion order; params pytree flattened with
    // dict keys sorted (JAX order): attn{W,b,beta,g}, attn_id{...},
    // cell[3]{W,b,beta,g}, cell_id[3]{...}, fc1{W,b,beta,g}, fc2{W,b,beta,g}.
    const float* sites = (const float*)d_in[0];

    StreamP SP[2];
    SP[0] = { (const float*)d_in[1], (const float*)d_in[3],
              (const float*)d_in[13], (const float*)d_in[14], (const float*)d_in[15], (const float*)d_in[16],
              (const float*)d_in[17], (const float*)d_in[18], (const float*)d_in[19], (const float*)d_in[20],
              (const float*)d_in[21], (const float*)d_in[22], (const float*)d_in[23], (const float*)d_in[24],
              (const float*)d_in[7],
              (const int*)d_in[45], (const int*)d_in[46], (const int*)d_in[47] };
    SP[1] = { (const float*)d_in[2], (const float*)d_in[4],
              (const float*)d_in[25], (const float*)d_in[26], (const float*)d_in[27], (const float*)d_in[28],
              (const float*)d_in[29], (const float*)d_in[30], (const float*)d_in[31], (const float*)d_in[32],
              (const float*)d_in[33], (const float*)d_in[34], (const float*)d_in[35], (const float*)d_in[36],
              (const float*)d_in[11],
              (const int*)d_in[48], (const int*)d_in[49], (const int*)d_in[50] };

    const float *fc1W = (const float*)d_in[37], *fc1b = (const float*)d_in[38],
                *fc1be = (const float*)d_in[39], *fc1g = (const float*)d_in[40];
    const float *fc2W = (const float*)d_in[41], *fc2b = (const float*)d_in[42],
                *fc2be = (const float*)d_in[43], *fc2g = (const float*)d_in[44];

    // Workspace carve (~170 MB)
    char* p = (char*)d_ws;
    auto alloc = [&](size_t bytes) -> void* {
        void* r = (void*)p; p += (bytes + 255) & ~(size_t)255; return r;
    };
    _Float16* W1t  = (_Float16*)alloc((size_t)GRp * D1 * FF * 2);
    _Float16* W2t  = (_Float16*)alloc((size_t)D1 * D1 * 2);
    _Float16* W3t  = (_Float16*)alloc((size_t)D3 * D1 * 2);
    _Float16* A1   = (_Float16*)alloc((size_t)MPAD * FF * 2);
    _Float16* BUF0 = (_Float16*)alloc((size_t)CHUNK * D1 * 2);
    _Float16* BUF1 = (_Float16*)alloc((size_t)CHUNK * D1 * 2);
    _Float16* Y3   = (_Float16*)alloc((size_t)CHUNK * D3 * 2);
    float*    upd  = (float*)alloc((size_t)NN * SS * 4);
    int* gOf       = (int*)alloc((size_t)EE * 4);
    int* counts    = (int*)alloc(16 * 4);
    int* starts    = (int*)alloc(16 * 4);
    int* cursor    = (int*)alloc(16 * 4);
    int* tileG     = (int*)alloc((size_t)MTILES * 4);
    int* edgeOfRow = (int*)alloc((size_t)MPAD * 4);
    int* rowOfEdge = (int*)alloc((size_t)EE * 4);

    auto gridOf = [](long long n) { return (unsigned)((n + 255) / 256); };

    fill_u32<<<gridOf((long long)NN * SS), 256, 0, stream>>>((uint32_t*)upd, 0u, (long long)NN * SS);

    for (int s = 0; s < 2; s++) {
        const StreamP& P = SP[s];
        // weight conversion / layout
        convert_w1<<<gridOf((long long)GRp * D1 * FF), 256, 0, stream>>>(P.w1, W1t);
        cast_f32_f16<<<gridOf((long long)D1 * D1), 256, 0, stream>>>(P.w2, W2t, (long long)D1 * D1);
        cast_f32_f16<<<gridOf((long long)D3 * D1), 256, 0, stream>>>(P.w3, W3t, (long long)D3 * D1);
        // bucket edges by group index
        fill_u32<<<1, 256, 0, stream>>>((uint32_t*)counts, 0u, 12);
        fill_u32<<<gridOf(MPAD), 256, 0, stream>>>((uint32_t*)edgeOfRow, 0xFFFFFFFFu, MPAD);
        fill_u32<<<gridOf((long long)MPAD * FF / 2), 256, 0, stream>>>(
            (uint32_t*)A1, 0u, (long long)MPAD * FF / 2);
        compute_g_hist<<<gridOf(EE), 256, 0, stream>>>(P.oh, gOf, counts);
        scan_tiles<<<1, 1, 0, stream>>>(counts, starts, tileG, cursor);
        scatter_rows<<<gridOf(EE), 256, 0, stream>>>(gOf, cursor, rowOfEdge, edgeOfRow);
        build_A1<<<EE, FF, 0, stream>>>(sites, P.i1, P.i2, P.bonds, rowOfEdge, A1);

        for (int c = 0; c < NCHUNK; c++) {
            const int m0 = c * CHUNK;
            // L1: per-bucket 160 -> 3072
            gemm_wmma_f16<<<dim3(D1 / 128, CHUNK / 128), 256, 0, stream>>>(
                A1 + (long long)m0 * FF, FF, W1t, BUF0, D1, FF,
                tileG, m0 / 128, (long long)D1 * FF);
            ln_rows<<<CHUNK, 256, 0, stream>>>(BUF0, D1, P.b1, P.g1, P.be1);
            // L2: 3072 -> 3072 (dominant GEMM)
            gemm_wmma_f16<<<dim3(D1 / 128, CHUNK / 128), 256, 0, stream>>>(
                BUF0, D1, W2t, BUF1, D1, D1, nullptr, 0, 0);
            ln_rows<<<CHUNK, 256, 0, stream>>>(BUF1, D1, P.b2, P.g2, P.be2);
            // L3: 3072 -> 768
            gemm_wmma_f16<<<dim3(D3 / 128, CHUNK / 128), 256, 0, stream>>>(
                BUF1, D1, W3t, Y3, D3, D1, nullptr, 0, 0);
            ln_rows<<<CHUNK, 256, 0, stream>>>(Y3, D3, P.b3, P.g3, P.be3);
            // gather + gate + segment-sum
            gate_scatter<<<dim3(CHUNK / 4), dim3(64, 4), 0, stream>>>(
                Y3, edgeOfRow, m0, P.i1c, P.i1, P.attnBeta, upd);
        }
    }

    node_update<<<NN, 64, 0, stream>>>(sites, upd, fc1W, fc1b, fc1g, fc1be,
                                       fc2W, fc2b, fc2g, fc2be, (float*)d_out);
}
